// CausalSelfAttention_23991687315661
// MI455X (gfx1250) — matrix operand order repaired and checked by reading, not compiled, not run on hardware
//
#include <hip/hip_runtime.h>
#include <hip/hip_bf16.h>

// ---------------------------------------------------------------------------
// CDNA5 (gfx1250) causal self-attention, bf16 WMMA pipeline, fp32 accumulate.
// GEMMs: 32x64 tile per wave (8 wmma / K-step). Attention: flash-style with
// async global->LDS staging of the K tile (ASYNCcnt) and LDS-transposed V.
// ---------------------------------------------------------------------------

typedef __bf16 bhalf;
typedef __attribute__((ext_vector_type(16))) __bf16 v16bf;
typedef __attribute__((ext_vector_type(8)))  __bf16 v8bf;
typedef __attribute__((ext_vector_type(8)))  float  v8f;

#define D_MODEL 1024
#define N_HEADS 16
#define DK      64
#define BATCH   4
#define SEQ     2048
#define BT      (BATCH * SEQ)      // 8192 rows
#define NQKV    (3 * D_MODEL)      // 3072 qkv channels

// fp32 -> bf16 with round-to-nearest-even (pure bit math, no native cvt dep)
__device__ __forceinline__ bhalf f2bf(float f) {
  unsigned u = __float_as_uint(f);
  unsigned r = (u + 0x7FFFu + ((u >> 16) & 1u)) >> 16;
  unsigned short s = (unsigned short)r;
  bhalf b;
  __builtin_memcpy(&b, &s, 2);
  return b;
}

// Generic pointer -> LDS byte offset (flat aperture: low 32 bits of the address).
__device__ __forceinline__ unsigned lds_off(const void* p) {
  return (unsigned)(uintptr_t)p;
}

// --- WMMA fragment loaders (ISA 7.12.2 layouts, wave32) ---------------------
// A-matrix 16x32 bf16: lane L: M = L%16, half h = L/16 selects K phase.
// Lane holds K = 16h..16h+15 (contiguous), matching the B fragment order.
__device__ __forceinline__ v16bf load_a_frag(const bhalf* base, int ld, int lane) {
  int m = lane & 15, h = lane >> 4;
  const bhalf* row = base + (size_t)m * ld;
  v8bf c0 = *(const v8bf*)(row + 16 * h);
  v8bf c1 = *(const v8bf*)(row + 16 * h + 8);
  v16bf a;
#pragma unroll
  for (int i = 0; i < 8; ++i) { a[i] = c0[i]; a[8 + i] = c1[i]; }
  return a;
}

// B-matrix 32x16 bf16: lane L: N = L%16, half h = L/16; lane holds contraction
// values K = 16h..16h+15 (contiguous) of column N. Source layout: element
// (k, n) lives at base[n*ld + k] (contraction dim contiguous per column).
__device__ __forceinline__ v16bf load_b_frag(const bhalf* base, int ld, int lane) {
  int n = lane & 15, h = lane >> 4;
  const bhalf* row = base + (size_t)n * ld + 16 * h;
  v8bf c0 = *(const v8bf*)(row);
  v8bf c1 = *(const v8bf*)(row + 8);
  v16bf b;
#pragma unroll
  for (int i = 0; i < 8; ++i) { b[i] = c0[i]; b[8 + i] = c1[i]; }
  return b;
}

// ---------------------------------------------------------------------------
// Kernel 1: fp32 -> bf16 conversion (grid-stride)
// ---------------------------------------------------------------------------
__global__ void cvt_bf16_kernel(const float* __restrict__ src,
                                bhalf* __restrict__ dst, int n) {
  int i = blockIdx.x * blockDim.x + threadIdx.x;
  int stride = gridDim.x * blockDim.x;
  for (; i < n; i += stride) dst[i] = f2bf(src[i]);
}

// ---------------------------------------------------------------------------
// Kernel 2: QKV projection GEMM (qkv = x @ W_qkv^T + b), scatter to Q/K/V in
// [B, H, T, DK] bf16 layout; softmax 1/sqrt(dk) folded into Q.
// Block = 128 threads = 4 waves; wave computes a 32x64 tile (8 wmma / K-step).
// ---------------------------------------------------------------------------
__global__ void qkv_gemm_kernel(const bhalf* __restrict__ X,
                                const bhalf* __restrict__ W,
                                const float* __restrict__ bias,
                                bhalf* __restrict__ Q,
                                bhalf* __restrict__ K,
                                bhalf* __restrict__ V) {
  const int lane = threadIdx.x & 31;
  const int wave = threadIdx.x >> 5;
  const int tn = blockIdx.x * 64;               // qkv channel tile (4x16)
  const int tm = blockIdx.y * 128 + wave * 32;  // row tile (2x16)

  const bhalf* arow0 = X + (size_t)tm * D_MODEL;
  const bhalf* arow1 = X + (size_t)(tm + 16) * D_MODEL;

  v8f acc[2][4] = {};
#pragma unroll 2
  for (int kk = 0; kk < D_MODEL; kk += 32) {
    __builtin_prefetch(arow0 + kk + 128, 0, 1);
    __builtin_prefetch(arow1 + kk + 128, 0, 1);
    v16bf af[2];
    af[0] = load_a_frag(arow0 + kk, D_MODEL, lane);
    af[1] = load_a_frag(arow1 + kk, D_MODEL, lane);
#pragma unroll
    for (int nt = 0; nt < 4; ++nt) {
      v16bf bf = load_b_frag(W + (size_t)(tn + nt * 16) * D_MODEL + kk,
                             D_MODEL, lane);
#pragma unroll
      for (int mt = 0; mt < 2; ++mt)
        acc[mt][nt] = __builtin_amdgcn_wmma_f32_16x16x32_bf16(
            false, af[mt], false, bf, (short)0, acc[mt][nt], false, false);
    }
  }

  // Epilogue: D layout lane L, VGPR r -> row = r + 8*(L/16), col = L%16.
#pragma unroll
  for (int nt = 0; nt < 4; ++nt) {
    const int n = tn + nt * 16 + (lane & 15);
    const int head = n / (3 * DK);
    const int r2 = n % (3 * DK);
    const float bias_n = bias[n];
#pragma unroll
    for (int mt = 0; mt < 2; ++mt) {
#pragma unroll
      for (int r = 0; r < 8; ++r) {
        int m = tm + mt * 16 + r + 8 * (lane >> 4);
        int bi = m / SEQ, t = m % SEQ;
        float v = acc[mt][nt][r] + bias_n;
        size_t o = (((size_t)bi * N_HEADS + head) * SEQ + t) * DK;
        if (r2 < DK)            Q[o + r2]          = f2bf(v * 0.125f);
        else if (r2 < 2 * DK)   K[o + r2 - DK]     = f2bf(v);
        else                    V[o + r2 - 2 * DK] = f2bf(v);
      }
    }
  }
}

// ---------------------------------------------------------------------------
// Kernel 3: flash-style causal attention. Block = 128 threads = 4 waves,
// 64 queries per block, wave owns 16 queries x full DK=64 output.
// Per 32-key step: K tile staged via GLOBAL_LOAD_ASYNC_TO_LDS_B128 (ASYNCcnt),
// V tile staged transposed via ds stores, 4 wmma for S = Q K^T, online softmax
// (fp32, shfl_xor row reductions within 16-lane halves), P re-laid out through
// a per-wave LDS tile, 4 wmma for P @ V.
// ---------------------------------------------------------------------------
__global__ void attn_kernel(const bhalf* __restrict__ Q,
                            const bhalf* __restrict__ K,
                            const bhalf* __restrict__ V,
                            bhalf* __restrict__ O) {
  __shared__ __align__(16) bhalf ldsK[32][DK];     // K tile [key][d] (async)
  __shared__ __align__(16) bhalf ldsV[DK][32];     // V tile transposed [d][key]
  __shared__ __align__(16) bhalf ldsP[4][16][32];  // per-wave P tile [m][key]

  const int tid  = threadIdx.x;
  const int lane = tid & 31;
  const int wave = tid >> 5;
  const int bh   = blockIdx.y;          // b*H + h
  const int qblk = blockIdx.x * 64;     // block's query base
  const int qtile = qblk + wave * 16;   // wave's 16 queries

  const bhalf* Qp = Q + (size_t)bh * SEQ * DK;
  const bhalf* Kp = K + (size_t)bh * SEQ * DK;
  const bhalf* Vp = V + (size_t)bh * SEQ * DK;

  // Q fragments for both dk chunks (contraction 0..31, 32..63), loaded once.
  v16bf aq0 = load_a_frag(Qp + (size_t)qtile * DK, DK, lane);
  v16bf aq1 = load_a_frag(Qp + (size_t)qtile * DK + 32, DK, lane);

  v8f acc[4];
#pragma unroll
  for (int nt = 0; nt < 4; ++nt) acc[nt] = {};
  float mrow[8], lrow[8];
#pragma unroll
  for (int r = 0; r < 8; ++r) { mrow[r] = -1e30f; lrow[r] = 0.f; }

  const unsigned ldsK_base = lds_off(&ldsK[0][0]);
  const int jend = qblk + 64;
  for (int j = 0; j < jend; j += 32) {
    __syncthreads();
    // Async-stage K tile (32 keys x 64 dk = 4 KB): each thread DMAs two 16B
    // chunks global->LDS without touching VGPR data. Tracked by ASYNCcnt.
#pragma unroll
    for (int i = 0; i < 2; ++i) {
      int c = i * 128 + tid;            // 16B chunk index, 0..255
      int key = c >> 3;                 // 8 chunks per 64-elem key row
      int d = (c & 7) * 8;
      const bhalf* gp = Kp + (size_t)(j + key) * DK + d;
      unsigned dst = ldsK_base + (unsigned)c * 16u;
      asm volatile("global_load_async_to_lds_b128 %0, %1, off"
                   :: "v"(dst), "v"((unsigned long long)(uintptr_t)gp)
                   : "memory");
    }
    // Stage V tile (32 keys x 64 dk) transposed into LDS (ds stores).
#pragma unroll
    for (int i = 0; i < 16; ++i) {
      int e = i * 128 + tid;            // 0..2047, d fastest -> coalesced
      int key = e >> 6;
      int d = e & 63;
      ldsV[d][key] = Vp[(size_t)(j + key) * DK + d];
    }
    asm volatile("s_wait_asynccnt 0x0" ::: "memory");
    __syncthreads();

    if (j <= qtile + 15) {              // wave-uniform causal early-out
      // S = Q K^T for two 16-key column tiles; K fragments from LDS.
      v8f S[2]; S[0] = {}; S[1] = {};
#pragma unroll
      for (int ct = 0; ct < 2; ++ct) {
        const bhalf* kb = &ldsK[ct * 16][0];
        v16bf b0 = load_b_frag(kb, DK, lane);
        v16bf b1 = load_b_frag(kb + 32, DK, lane);
        S[ct] = __builtin_amdgcn_wmma_f32_16x16x32_bf16(false, aq0, false, b0,
                                                        (short)0, S[ct], false, false);
        S[ct] = __builtin_amdgcn_wmma_f32_16x16x32_bf16(false, aq1, false, b1,
                                                        (short)0, S[ct], false, false);
      }

      // Causal mask + online softmax. C/D layout: lane holds
      // (row = r + 8*(L/16), col = L%16); row reduction = xor-shuffle over the
      // 16 lanes of the half (masks 1,2,4,8 never cross the half boundary).
      const int qrow_base = qtile + 8 * (lane >> 4);
      const int kcol = lane & 15;
#pragma unroll
      for (int r = 0; r < 8; ++r) {
        int qg = qrow_base + r;
        float s0 = ((j + kcol) <= qg) ? S[0][r] : -1e30f;
        float s1 = ((j + 16 + kcol) <= qg) ? S[1][r] : -1e30f;
        float mx = fmaxf(s0, s1);
        mx = fmaxf(mx, __shfl_xor(mx, 1, 32));
        mx = fmaxf(mx, __shfl_xor(mx, 2, 32));
        mx = fmaxf(mx, __shfl_xor(mx, 4, 32));
        mx = fmaxf(mx, __shfl_xor(mx, 8, 32));
        float mnew = fmaxf(mrow[r], mx);
        float sc = __expf(mrow[r] - mnew);
        mrow[r] = mnew;
        float p0 = __expf(s0 - mnew);
        float p1 = __expf(s1 - mnew);
        float rs = p0 + p1;
        rs += __shfl_xor(rs, 1, 32);
        rs += __shfl_xor(rs, 2, 32);
        rs += __shfl_xor(rs, 4, 32);
        rs += __shfl_xor(rs, 8, 32);
        lrow[r] = lrow[r] * sc + rs;
        // Stash P into the wave-private LDS tile to re-layout as A-fragment.
        int m = r + 8 * (lane >> 4);
        ldsP[wave][m][kcol]      = f2bf(p0);
        ldsP[wave][m][16 + kcol] = f2bf(p1);
        // Rescale running accumulators (WMMA then adds P@V on top).
#pragma unroll
        for (int nt = 0; nt < 4; ++nt) acc[nt][r] *= sc;
      }

      // acc += P (16x32) @ Vtile (32x64). LDS ops are in-order per wave, so
      // the A-fragment read below observes the stores above.
      v16bf ap = load_a_frag(&ldsP[wave][0][0], 32, lane);
#pragma unroll
      for (int nt = 0; nt < 4; ++nt) {
        v16bf bv = load_b_frag(&ldsV[nt * 16][0], 32, lane);
        acc[nt] = __builtin_amdgcn_wmma_f32_16x16x32_bf16(false, ap, false, bv,
                                                          (short)0, acc[nt], false, false);
      }
    }
  }

  // Normalize and write attention output in [B, T, C] bf16 layout.
  const int b = bh / N_HEADS, h = bh % N_HEADS;
  const int col = lane & 15;
#pragma unroll
  for (int r = 0; r < 8; ++r) {
    int t = qtile + r + 8 * (lane >> 4);
    float inv = 1.0f / lrow[r];
    size_t o = ((size_t)b * SEQ + t) * D_MODEL + h * DK;
#pragma unroll
    for (int nt = 0; nt < 4; ++nt)
      O[o + nt * 16 + col] = f2bf(acc[nt][r] * inv);
  }
}

// ---------------------------------------------------------------------------
// Kernel 4: output projection GEMM (out = attn @ W_out^T + b_out), fp32 out.
// Same 32x64-per-wave blocking as kernel 2.
// ---------------------------------------------------------------------------
__global__ void out_gemm_kernel(const bhalf* __restrict__ A,
                                const bhalf* __restrict__ W,
                                const float* __restrict__ bias,
                                float* __restrict__ out) {
  const int lane = threadIdx.x & 31;
  const int wave = threadIdx.x >> 5;
  const int tn = blockIdx.x * 64;
  const int tm = blockIdx.y * 128 + wave * 32;

  const bhalf* arow0 = A + (size_t)tm * D_MODEL;
  const bhalf* arow1 = A + (size_t)(tm + 16) * D_MODEL;

  v8f acc[2][4] = {};
#pragma unroll 2
  for (int kk = 0; kk < D_MODEL; kk += 32) {
    __builtin_prefetch(arow0 + kk + 128, 0, 1);
    __builtin_prefetch(arow1 + kk + 128, 0, 1);
    v16bf af[2];
    af[0] = load_a_frag(arow0 + kk, D_MODEL, lane);
    af[1] = load_a_frag(arow1 + kk, D_MODEL, lane);
#pragma unroll
    for (int nt = 0; nt < 4; ++nt) {
      v16bf bf = load_b_frag(W + (size_t)(tn + nt * 16) * D_MODEL + kk,
                             D_MODEL, lane);
#pragma unroll
      for (int mt = 0; mt < 2; ++mt)
        acc[mt][nt] = __builtin_amdgcn_wmma_f32_16x16x32_bf16(
            false, af[mt], false, bf, (short)0, acc[mt][nt], false, false);
    }
  }

#pragma unroll
  for (int nt = 0; nt < 4; ++nt) {
    const int n = tn + nt * 16 + (lane & 15);
    const float bias_n = bias[n];
#pragma unroll
    for (int mt = 0; mt < 2; ++mt) {
#pragma unroll
      for (int r = 0; r < 8; ++r) {
        int m = tm + mt * 16 + r + 8 * (lane >> 4);
        out[(size_t)m * D_MODEL + n] = acc[mt][nt][r] + bias_n;
      }
    }
  }
}

// ---------------------------------------------------------------------------
// Host-side launcher. Workspace layout (bf16 buffers, all 16B aligned):
//   xb [BT,C] 16MB | Wqkvb [3C,C] 6MB | Woutb [C,C] 2MB |
//   Qb,Kb,Vb [B,H,T,DK] 16MB each | attnb [BT,C] 16MB   (~88 MB total)
// ---------------------------------------------------------------------------
extern "C" void kernel_launch(void* const* d_in, const int* in_sizes, int n_in,
                              void* d_out, int out_size, void* d_ws, size_t ws_size,
                              hipStream_t stream) {
  const float* x     = (const float*)d_in[0];
  const float* W_qkv = (const float*)d_in[1];
  const float* b_qkv = (const float*)d_in[2];
  const float* W_out = (const float*)d_in[3];
  const float* b_out = (const float*)d_in[4];
  float* out = (float*)d_out;

  char* ws = (char*)d_ws;
  const size_t SZ_X    = (size_t)BT * D_MODEL * sizeof(bhalf);
  const size_t SZ_WQKV = (size_t)NQKV * D_MODEL * sizeof(bhalf);
  const size_t SZ_WOUT = (size_t)D_MODEL * D_MODEL * sizeof(bhalf);
  const size_t SZ_HEAD = (size_t)BATCH * N_HEADS * SEQ * DK * sizeof(bhalf);

  bhalf* xb    = (bhalf*)(ws);
  bhalf* Wqkvb = (bhalf*)(ws + SZ_X);
  bhalf* Woutb = (bhalf*)(ws + SZ_X + SZ_WQKV);
  bhalf* Qb    = (bhalf*)(ws + SZ_X + SZ_WQKV + SZ_WOUT);
  bhalf* Kb    = (bhalf*)(ws + SZ_X + SZ_WQKV + SZ_WOUT + SZ_HEAD);
  bhalf* Vb    = (bhalf*)(ws + SZ_X + SZ_WQKV + SZ_WOUT + 2 * SZ_HEAD);
  bhalf* attnb = (bhalf*)(ws + SZ_X + SZ_WQKV + SZ_WOUT + 3 * SZ_HEAD);

  cvt_bf16_kernel<<<512, 256, 0, stream>>>(x, xb, BT * D_MODEL);
  cvt_bf16_kernel<<<512, 256, 0, stream>>>(W_qkv, Wqkvb, NQKV * D_MODEL);
  cvt_bf16_kernel<<<256, 256, 0, stream>>>(W_out, Woutb, D_MODEL * D_MODEL);

  qkv_gemm_kernel<<<dim3(NQKV / 64, BT / 128), 128, 0, stream>>>(
      xb, Wqkvb, b_qkv, Qb, Kb, Vb);

  attn_kernel<<<dim3(SEQ / 64, BATCH * N_HEADS), 128, 0, stream>>>(
      Qb, Kb, Vb, attnb);

  out_gemm_kernel<<<dim3(D_MODEL / 64, BT / 128), 128, 0, stream>>>(
      attnb, Woutb, b_out, out);
}
